// FusionBlock_85839216377993
// MI455X (gfx1250) — compile-verified, matrix operand order checked by reading, not run on hardware
//
#include <hip/hip_runtime.h>
#include <hip/hip_bf16.h>

// ---------------------------------------------------------------------------
// Problem constants
// ---------------------------------------------------------------------------
#define BB   8
#define CIN  16
#define COUT 16
#define TT   32
#define HH   16
#define WW   16
#define NH   4
#define DIN  4096          // CIN*H*W
#define DOUT 4096          // COUT*H*W
#define DH   1024          // DOUT/NH
#define MROWS (BB*TT)      // 256

typedef __attribute__((ext_vector_type(16))) _Float16 v16h;
typedef __attribute__((ext_vector_type(8)))  float    v8f;

union FragU {
  v16h h;
  uint4 q[2];
};

// ---------------------------------------------------------------------------
// WMMA 16-bit A/B fragment layout (CDNA5 ISA §7.12.2):
//   element e of per-lane v16h holds K = (e&7) + ((e>>3)<<4) + ((lane>>4)<<3)
//   i.e. two contiguous 16B chunks at K-offsets {0,+16} (plus +8 for hi half-
//   wave).  Row (A) / column (B) = lane & 15.  Matrices row-major K-contig.
// ---------------------------------------------------------------------------
__device__ __forceinline__ v16h frag_load_g(const _Float16* __restrict__ base,
                                            int ld, int row, int k0, int lane) {
  const uint4* p = (const uint4*)(base + (size_t)row * ld + k0 + ((lane >> 4) << 3));
  FragU u;
  u.q[0] = p[0];   // K offsets 0..7   (halves)
  u.q[1] = p[2];   // K offsets 16..23 (halves)
  return u.h;
}

__device__ __forceinline__ v16h frag_zero() {
  v16h f;
#pragma unroll
  for (int e = 0; e < 16; e++) f[e] = (_Float16)0.0f;
  return f;
}

// ---------------------------------------------------------------------------
// gfx1250 async global->LDS copy (GLOBAL_LOAD_ASYNC_TO_LDS_B128, ASYNCcnt).
// Per-lane: copies 16B from the lane's global address to the lane's LDS
// address.  GV addressing mode (saddr = off).  Generic LDS pointers keep the
// LDS offset in addr[31:0] (CDNA5 ISA aperture mapping).
// ---------------------------------------------------------------------------
__device__ __forceinline__ void async_copy_b128(void* lds_dst, const void* gsrc) {
  unsigned l = (unsigned)(uintptr_t)lds_dst;
  asm volatile("global_load_async_to_lds_b128 %0, %1, off"
               :: "v"(l), "v"(gsrc)
               : "memory");
}
__device__ __forceinline__ void wait_async0() {
  asm volatile("s_wait_asynccnt 0x0" ::: "memory");
}

// ---------------------------------------------------------------------------
// GEMM:  C[m,n] = sum_k A[m,k] * Bw[n,k] + bias[n]      (f16 in, f32 out)
//
// Block = 128 thr (4 waves).  Block tile 64M x 256N; wave tile 64M x 64N
// (4x4 accumulators).  A k-slab (64 rows x 64 K, f16) is staged in double-
// buffered LDS (row stride padded to 72 halves for bank spread) via async
// global->LDS copies and shared by all 4 waves; B streams straight from
// global (L2-resident) as b128 pairs.  Each k-step: 8 fragment loads feeding
// 16 independent WMMAs.  One barrier per slab: async fills for slab s+1 are
// issued at loop top (previous barrier already closed all reads of that
// buffer), and the bottom wait_asynccnt+barrier publishes them.
// grid.x = N/256, grid.y = M/64.  M%64==0, N%256==0, K%64==0 for all uses.
// ---------------------------------------------------------------------------
#define KSLAB 64
#define APAD  72

__global__ void __launch_bounds__(128)
k_gemm_f16(const _Float16* __restrict__ A,
           const _Float16* __restrict__ Bw,
           const float*    __restrict__ bias,
           float*          __restrict__ C,
           int M, int N, int K) {
  __shared__ _Float16 As[2][64][APAD];

  int tid    = threadIdx.x;
  int lane   = tid & 31;
  int wave   = tid >> 5;                    // 0..3
  int m_blk  = blockIdx.y << 6;             // 64 rows per block
  int n_base = (blockIdx.x << 8) + (wave << 6);
  int col    = lane & 15;
  int hiofs  = (lane >> 4) << 3;            // +8 halves for upper half-wave

  // this thread's four 16B staging chunks: chunk id = tid + 128*j
  int srow[4], scol[4];
#pragma unroll
  for (int j = 0; j < 4; j++) {
    int id  = tid + 128 * j;                // 512 chunks of 16B per slab
    srow[j] = id >> 3;
    scol[j] = (id & 7) * 8;                 // halves
  }

  v8f acc[4][4];
#pragma unroll
  for (int i = 0; i < 4; i++)
#pragma unroll
    for (int j = 0; j < 4; j++) acc[i][j] = (v8f){};

  const int nslab = K / KSLAB;

  // ---- prologue: async-stage slab 0 into LDS buffer 0 -------------------
#pragma unroll
  for (int j = 0; j < 4; j++)
    async_copy_b128(&As[0][srow[j]][scol[j]],
                    A + (size_t)(m_blk + srow[j]) * K + scol[j]);
  wait_async0();
  __syncthreads();

  for (int s = 0; s < nslab; s++) {
    int cur = s & 1;

    // ---- issue async fills for next slab (buffer cur^1 is free: the
    //      previous barrier closed all reads of it) ----------------------
    if (s + 1 < nslab) {
      int k0n = (s + 1) * KSLAB;
#pragma unroll
      for (int j = 0; j < 4; j++)
        async_copy_b128(&As[cur ^ 1][srow[j]][scol[j]],
                        A + (size_t)(m_blk + srow[j]) * K + k0n + scol[j]);
    }

    // ---- compute on current slab: 2 k-steps of 32 ----------------------
#pragma unroll
    for (int ks = 0; ks < KSLAB; ks += 32) {
      int kg = s * KSLAB + ks;

      v16h af[4];
#pragma unroll
      for (int i = 0; i < 4; i++) {
        const uint4* p = (const uint4*)&As[cur][i * 16 + col][ks + hiofs];
        FragU u; u.q[0] = p[0]; u.q[1] = p[2];
        af[i] = u.h;
      }
      v16h bf[4];
#pragma unroll
      for (int j = 0; j < 4; j++)
        bf[j] = frag_load_g(Bw, K, n_base + j * 16 + col, kg, lane);

#pragma unroll
      for (int i = 0; i < 4; i++)
#pragma unroll
        for (int j = 0; j < 4; j++)
          acc[i][j] = __builtin_amdgcn_wmma_f32_16x16x32_f16(
              false, af[i], false, bf[j], (short)0, acc[i][j], false, false);
    }

    // my async fills landed; barrier publishes them to all waves and also
    // closes this slab's reads before anyone refills buffer `cur`.
    wait_async0();
    __syncthreads();
  }

  // ---- epilogue: bias + store ------------------------------------------
#pragma unroll
  for (int i = 0; i < 4; i++) {
#pragma unroll
    for (int r = 0; r < 8; r++) {
      int m = m_blk + i * 16 + r + ((lane >> 4) << 3);
      float* crow = C + (size_t)m * N;
#pragma unroll
      for (int j = 0; j < 4; j++) {
        int n = n_base + j * 16 + col;
        crow[n] = acc[i][j][r] + bias[n];
      }
    }
  }
}

// ---------------------------------------------------------------------------
// sLSTM step t.  grid = (DH/16, NH), block = 128 (wave w == gate g).
// Each wave: 16x16 WMMA tile over K=1024 against Rg[head][g]; rows >= 8 of A
// (batch dim) zero.  Gate tile staged in LDS, then stabilized gate math.
// h double-buffered across launches (stream order gives the t->t+1 dep).
// Rg in f16 (33.5 MB) stays resident in the 192 MB L2 across all 32 steps.
// ---------------------------------------------------------------------------
__global__ void k_lstm_step(const _Float16* __restrict__ h_prev,  // [8][4096]
                            const _Float16* __restrict__ Rg,      // [4][4][1024][1024]
                            const float*    __restrict__ Gx,      // [8][32][4][4096]
                            float* __restrict__ cst,
                            float* __restrict__ nst,
                            float* __restrict__ mst,              // [8][4096]
                            float* __restrict__ hs,               // [8][32][4096]
                            _Float16* __restrict__ h_next,        // [8][4096]
                            int t) {
  __shared__ float gh[4][8][16];
  int lane  = threadIdx.x & 31;
  int g     = threadIdx.x >> 5;        // gate = wave
  int head  = blockIdx.y;
  int otile = blockIdx.x << 4;
  int col   = lane & 15;

  const _Float16* Bbase = Rg + (size_t)(head * 4 + g) * DH * DH;
  v8f acc = {};
  for (int k0 = 0; k0 < DH; k0 += 32) {
    v16h a = (col < BB) ? frag_load_g(h_prev, DOUT, col, head * DH + k0, lane)
                        : frag_zero();
    v16h b = frag_load_g(Bbase, DH, otile + col, k0, lane);
    acc = __builtin_amdgcn_wmma_f32_16x16x32_f16(false, a, false, b, (short)0, acc, false, false);
  }
#pragma unroll
  for (int r = 0; r < 8; r++) {
    int b = r + ((lane >> 4) << 3);
    if (b < BB) gh[g][b][col] = acc[r];
  }
  __syncthreads();

  // gate math: 128 threads -> (b, o_local)
  int b  = threadIdx.x >> 4;           // 0..7
  int ol = threadIdx.x & 15;
  int d  = head * DH + otile + ol;
  size_t s      = (size_t)b * DOUT + d;
  size_t gxbase = ((size_t)(b * TT + t) * 4) * DOUT + d;

  float ipre = Gx[gxbase + 0ull * DOUT] + gh[0][b][ol];
  float fpre = Gx[gxbase + 1ull * DOUT] + gh[1][b][ol];
  float zpre = Gx[gxbase + 2ull * DOUT] + gh[2][b][ol];
  float opre = Gx[gxbase + 3ull * DOUT] + gh[3][b][ol];

  float mo = mst[s];
  float mn = fmaxf(fpre + mo, ipre);
  float iv = __expf(ipre - mn);
  float fv = __expf(fpre + mo - mn);
  float cv = fv * cst[s] + iv * tanhf(zpre);
  float nv = fv * nst[s] + iv;
  float hv = (1.0f / (1.0f + __expf(-opre))) * cv / nv;

  cst[s] = cv; nst[s] = nv; mst[s] = mn;
  hs[(size_t)(b * TT + t) * DOUT + d] = hv;
  h_next[s] = (_Float16)hv;
}

// ---------------------------------------------------------------------------
// Plain 3x3x3 conv, pad 1.  One thread per output element (16-ch: 2.7 GF
// total, not worth an implicit-GEMM path).  Layout NCDHW.
// ---------------------------------------------------------------------------
__global__ void k_conv3d(const float* __restrict__ x, const float* __restrict__ w,
                         const float* __restrict__ bias, float* __restrict__ y,
                         int relu) {
  int idx = blockIdx.x * blockDim.x + threadIdx.x;
  if (idx >= BB * COUT * TT * HH * WW) return;
  int wi = idx & 15, hi = (idx >> 4) & 15, ti = (idx >> 8) & 31;
  int co = (idx >> 13) & 15, b = idx >> 17;
  float acc = bias[co];
  for (int ci = 0; ci < CIN; ci++) {
#pragma unroll
    for (int dt = -1; dt <= 1; dt++) {
      int tt = ti + dt; if ((unsigned)tt >= TT) continue;
#pragma unroll
      for (int dh = -1; dh <= 1; dh++) {
        int hh = hi + dh; if ((unsigned)hh >= HH) continue;
#pragma unroll
        for (int dw = -1; dw <= 1; dw++) {
          int ww = wi + dw; if ((unsigned)ww >= WW) continue;
          acc += x[(((b * CIN + ci) * TT + tt) * HH + hh) * WW + ww] *
                 w[(((co * CIN + ci) * 3 + (dt + 1)) * 3 + (dh + 1)) * 3 + (dw + 1)];
        }
      }
    }
  }
  y[idx] = relu ? fmaxf(acc, 0.0f) : acc;
}

// Instance norm + relu.  One block per (b,c); reduce 8192 elements.
__global__ void k_instnorm_relu(const float* __restrict__ y, float* __restrict__ o) {
  __shared__ float ssum[256], ssq[256];
  const float* p = y + (size_t)blockIdx.x * 8192;
  float s = 0.0f, q = 0.0f;
  for (int i = threadIdx.x; i < 8192; i += 256) { float v = p[i]; s += v; q += v * v; }
  ssum[threadIdx.x] = s; ssq[threadIdx.x] = q;
  __syncthreads();
  for (int st = 128; st > 0; st >>= 1) {
    if ((int)threadIdx.x < st) {
      ssum[threadIdx.x] += ssum[threadIdx.x + st];
      ssq[threadIdx.x]  += ssq[threadIdx.x + st];
    }
    __syncthreads();
  }
  float mu  = ssum[0] * (1.0f / 8192.0f);
  float var = ssq[0] * (1.0f / 8192.0f) - mu * mu;
  float rs  = rsqrtf(var + 1e-5f);
  float* po = o + (size_t)blockIdx.x * 8192;
  for (int i = threadIdx.x; i < 8192; i += 256) po[i] = fmaxf((p[i] - mu) * rs, 0.0f);
}

// f32 -> f16 convert
__global__ void k_f32_to_f16(const float* __restrict__ a, _Float16* __restrict__ o, int n) {
  int i = blockIdx.x * blockDim.x + threadIdx.x;
  if (i < n) o[i] = (_Float16)a[i];
}

// x (B,CIN,T,H,W) -> Xh (B*T, CIN*H*W) in f16
__global__ void k_pack_x(const float* __restrict__ x, _Float16* __restrict__ o) {
  int idx = blockIdx.x * blockDim.x + threadIdx.x;
  if (idx >= BB * TT * DIN) return;
  int d = idx & 4095, t = (idx >> 12) & 31, b = idx >> 17;
  int ci = d >> 8, rest = d & 255;
  o[idx] = (_Float16)x[(((b * CIN + ci) * TT + t) << 8) + rest];
}

// Final sigmoid-attention blend.  One thread per (b,t,h,w) position.
__global__ void k_combine(const float* __restrict__ hs, const float* __restrict__ out3d,
                          const float* __restrict__ aw, const float* __restrict__ ab,
                          float* __restrict__ out) {
  int idx = blockIdx.x * blockDim.x + threadIdx.x;
  if (idx >= BB * TT * HH * WW) return;
  int pos = idx & 255, t = (idx >> 8) & 31, b = idx >> 13;
  float acc = ab[0];
  float sv[COUT], ov[COUT];
#pragma unroll
  for (int c = 0; c < COUT; c++) {
    float s  = hs[(size_t)(b * TT + t) * DOUT + c * 256 + pos];
    float o3 = out3d[((size_t)((b * COUT + c) * TT + t) << 8) + pos];
    sv[c] = s; ov[c] = o3;
    acc += aw[c] * s + aw[COUT + c] * o3;
  }
  float alpha = 1.0f / (1.0f + __expf(-acc));
#pragma unroll
  for (int c = 0; c < COUT; c++) {
    out[((size_t)((b * COUT + c) * TT + t) << 8) + pos] =
        alpha * ov[c] + (1.0f - alpha) * sv[c];
  }
}

// ---------------------------------------------------------------------------
extern "C" void kernel_launch(void* const* d_in, const int* in_sizes, int n_in,
                              void* d_out, int out_size, void* d_ws, size_t ws_size,
                              hipStream_t stream) {
  const float* x      = (const float*)d_in[0];
  const float* c1_w   = (const float*)d_in[1];
  const float* c1_b   = (const float*)d_in[2];
  const float* c2_w   = (const float*)d_in[3];
  const float* c2_b   = (const float*)d_in[4];
  const float* c3_w   = (const float*)d_in[5];
  const float* c3_b   = (const float*)d_in[6];
  const float* pre_w  = (const float*)d_in[7];
  const float* pre_b  = (const float*)d_in[8];
  const float* lstm_W = (const float*)d_in[9];
  const float* lstm_b = (const float*)d_in[10];
  const float* lstm_R = (const float*)d_in[11];
  const float* attn_w = (const float*)d_in[12];
  const float* attn_b = (const float*)d_in[13];
  float* out = (float*)d_out;

  char* ws = (char*)d_ws;
  size_t off = 0;
  auto alloc = [&](size_t bytes) -> void* {
    off = (off + 255) & ~(size_t)255;
    void* p = ws + off;
    off += bytes;
    return p;
  };

  const size_t nWpre = (size_t)DOUT * DIN;               // 16.7M
  const size_t nWg   = (size_t)4 * DOUT * DOUT;          // 67.1M
  const size_t nRg   = (size_t)NH * 4 * DH * DH;         // 16.7M
  const size_t nAct  = (size_t)MROWS * DIN;              // 1.05M
  const size_t nVox  = (size_t)BB * COUT * TT * HH * WW; // 1.05M

  _Float16* Wpre_h = (_Float16*)alloc(nWpre * 2);
  _Float16* Wg_h   = (_Float16*)alloc(nWg * 2);
  _Float16* Rg_h   = (_Float16*)alloc(nRg * 2);
  _Float16* Xh     = (_Float16*)alloc(nAct * 2);
  float*    xs     = (float*)   alloc(nAct * 4);
  _Float16* xs_h   = (_Float16*)alloc(nAct * 2);
  float*    Gx     = (float*)   alloc((size_t)MROWS * 4 * DOUT * 4);
  float*    y1     = (float*)   alloc(nVox * 4);
  float*    y1n    = (float*)   alloc(nVox * 4);
  float*    y2     = (float*)   alloc(nVox * 4);
  float*    out3d  = (float*)   alloc(nVox * 4);
  float*    cst    = (float*)   alloc((size_t)BB * DOUT * 4);
  float*    nst    = (float*)   alloc((size_t)BB * DOUT * 4);
  float*    mst    = (float*)   alloc((size_t)BB * DOUT * 4);
  _Float16* hA     = (_Float16*)alloc((size_t)BB * DOUT * 2);
  _Float16* hB     = (_Float16*)alloc((size_t)BB * DOUT * 2);
  float*    hs     = (float*)   alloc((size_t)BB * TT * DOUT * 4);

  // deterministic state init every launch (ws is poisoned, never re-zeroed)
  hipMemsetAsync(cst, 0, (size_t)BB * DOUT * 4, stream);
  hipMemsetAsync(nst, 0, (size_t)BB * DOUT * 4, stream);
  hipMemsetAsync(mst, 0, (size_t)BB * DOUT * 4, stream);
  hipMemsetAsync(hA,  0, (size_t)BB * DOUT * 2, stream);

  // weight conversion (one HBM sweep; halves all later weight traffic)
  k_f32_to_f16<<<(nWpre + 255) / 256, 256, 0, stream>>>(pre_w,  Wpre_h, (int)nWpre);
  k_f32_to_f16<<<(nWg   + 255) / 256, 256, 0, stream>>>(lstm_W, Wg_h,   (int)nWg);
  k_f32_to_f16<<<(nRg   + 255) / 256, 256, 0, stream>>>(lstm_R, Rg_h,   (int)nRg);
  k_pack_x    <<<(nAct  + 255) / 256, 256, 0, stream>>>(x, Xh);

  // xs = Xh @ pre_w^T + pre_b       (M=256, N=4096, K=4096)
  {
    dim3 g(DOUT / 256, MROWS / 64);
    k_gemm_f16<<<g, 128, 0, stream>>>(Xh, Wpre_h, pre_b, xs, MROWS, DOUT, DIN);
  }
  k_f32_to_f16<<<(nAct + 255) / 256, 256, 0, stream>>>(xs, xs_h, (int)nAct);

  // Gx = xs @ lstm_W^T + lstm_b     (M=256, N=16384, K=4096)
  {
    dim3 g((4 * DOUT) / 256, MROWS / 64);
    k_gemm_f16<<<g, 128, 0, stream>>>(xs_h, Wg_h, lstm_b, Gx, MROWS, 4 * DOUT, DOUT);
  }

  // conv branch (independent of the GEMM chain)
  k_conv3d<<<(nVox + 255) / 256, 256, 0, stream>>>(x,   c1_w, c1_b, y1,   0);
  k_instnorm_relu<<<BB * COUT, 256, 0, stream>>>(y1, y1n);
  k_conv3d<<<(nVox + 255) / 256, 256, 0, stream>>>(y1n, c2_w, c2_b, y2,   1);
  k_conv3d<<<(nVox + 255) / 256, 256, 0, stream>>>(y2,  c3_w, c3_b, out3d, 1);

  // sequential sLSTM recurrence: 32 launches, Rg_h L2-resident
  _Float16* hp = hA;
  _Float16* hn = hB;
  for (int t = 0; t < TT; t++) {
    dim3 g(DH / 16, NH);
    k_lstm_step<<<g, 128, 0, stream>>>(hp, Rg_h, Gx, cst, nst, mst, hs, hn, t);
    _Float16* tmp = hp; hp = hn; hn = tmp;
  }

  k_combine<<<(BB * TT * HH * WW + 255) / 256, 256, 0, stream>>>(hs, out3d, attn_w, attn_b, out);
}
